// SANLayer_5016521802481
// MI455X (gfx1250) — matrix operand order repaired and checked by reading, not compiled, not run on hardware
//
#include <hip/hip_runtime.h>
#include <hip/hip_bf16.h>
#include <stdint.h>

// ---------------------------------------------------------------------------
// SAN layer for MI455X (gfx1250, wave32, WMMA).
// Dominant cost: alpha @ L (two 4096^3 GEMMs, 275 GFLOP) done in f16 WMMA with
// f32 accumulation. Attention scores folded algebraically into 4 CIN-vectors
// + 4 scalars. All GEMM shapes are compile-time; tile fills are 128-bit and
// software-pipelined against the WMMA stage. Everything hot fits in 192MB L2.
// ---------------------------------------------------------------------------

#define NNODES 4096
#define CINC   128
#define COUTC  128
#define ATTC   256   // J*COUT
#define SLOPE  0.2f
#define NEGV  -9.0e15f

typedef __attribute__((ext_vector_type(16))) _Float16 v16h;
typedef __attribute__((ext_vector_type(8)))  _Float16 v8h;
typedef __attribute__((ext_vector_type(8)))  float    v8f;
typedef __attribute__((ext_vector_type(4)))  float    f4;

// ---------------------------------------------------------------------------
// k_prep: fold attention vectors into weights.
// vbuf: [0:128) v_irr1, [128:256) v_irr2, [256:384) v_sol1, [384:512) v_sol2,
//       [512..515] scalars b_irr1, b_irr2, b_sol1, b_sol2
// ---------------------------------------------------------------------------
__global__ __launch_bounds__(256) void k_prep(
    const float* __restrict__ Wi_w, const float* __restrict__ Wi_b,
    const float* __restrict__ Ws_w, const float* __restrict__ Ws_b,
    const float* __restrict__ att_irr, const float* __restrict__ att_sol,
    float* __restrict__ vbuf) {
  __shared__ float sred[256];
  const int t = threadIdx.x;

  if (t < CINC) {
    float v1 = 0.f, v2 = 0.f, v3 = 0.f, v4 = 0.f;
    for (int j = 0; j < 2; ++j) {
      for (int o = 0; o < COUTC; ++o) {
        const float wi = Wi_w[((size_t)j * CINC + t) * COUTC + o];
        const float ws = Ws_w[((size_t)j * CINC + t) * COUTC + o];
        const int a = j * COUTC + o;
        v1 += wi * att_irr[a];
        v2 += wi * att_irr[ATTC + a];
        v3 += ws * att_sol[a];
        v4 += ws * att_sol[ATTC + a];
      }
    }
    vbuf[t]       = v1;
    vbuf[128 + t] = v2;
    vbuf[256 + t] = v3;
    vbuf[384 + t] = v4;
  }

  float b[4];
  b[0] = Wi_b[t] * att_irr[t];
  b[1] = Wi_b[t] * att_irr[ATTC + t];
  b[2] = Ws_b[t] * att_sol[t];
  b[3] = Ws_b[t] * att_sol[ATTC + t];
  for (int s = 0; s < 4; ++s) {
    __syncthreads();
    sred[t] = b[s];
    __syncthreads();
    for (int off = 128; off > 0; off >>= 1) {
      if (t < off) sred[t] += sred[t + off];
      __syncthreads();
    }
    if (t == 0) vbuf[512 + s] = sred[0];
  }
}

// ---------------------------------------------------------------------------
// k_scores: sbuf[q*N + i] = dot(x[i,:], v_q) + b_q
// ---------------------------------------------------------------------------
__global__ __launch_bounds__(256) void k_scores(
    const float* __restrict__ x, const float* __restrict__ vbuf,
    float* __restrict__ sbuf) {
  __shared__ float sv[516];
  const int t = threadIdx.x;
  for (int i = t; i < 516; i += 256) sv[i] = vbuf[i];
  __syncthreads();

  const int row = blockIdx.x * 256 + t;
  const float* xr = x + (size_t)row * CINC;
  float a1 = 0.f, a2 = 0.f, a3 = 0.f, a4 = 0.f;
#pragma unroll 4
  for (int c = 0; c < CINC; ++c) {
    const float xv = xr[c];
    a1 += xv * sv[c];
    a2 += xv * sv[128 + c];
    a3 += xv * sv[256 + c];
    a4 += xv * sv[384 + c];
  }
  sbuf[row]              = a1 + sv[512];
  sbuf[NNODES + row]     = a2 + sv[513];
  sbuf[2 * NNODES + row] = a3 + sv[514];
  sbuf[3 * NNODES + row] = a4 + sv[515];
}

// ---------------------------------------------------------------------------
// k_cvt: f32 -> f16 (vectorized x4)
// ---------------------------------------------------------------------------
__global__ __launch_bounds__(256) void k_cvt(const float* __restrict__ in,
                                             _Float16* __restrict__ out, int n4) {
  const int i = blockIdx.x * 256 + threadIdx.x;
  if (i < n4) {
    const f4 v = ((const f4*)in)[i];
    _Float16 h0 = (_Float16)v.x, h1 = (_Float16)v.y;
    _Float16 h2 = (_Float16)v.z, h3 = (_Float16)v.w;
    ushort2 p0, p1;
    p0.x = *(const uint16_t*)&h0; p0.y = *(const uint16_t*)&h1;
    p1.x = *(const uint16_t*)&h2; p1.y = *(const uint16_t*)&h3;
    ((ushort2*)out)[2 * i]     = p0;
    ((ushort2*)out)[2 * i + 1] = p1;
  }
}

// ---------------------------------------------------------------------------
// k_softmax: one block per row; masked leaky scores -> f16 alpha row.
// ---------------------------------------------------------------------------
__global__ __launch_bounds__(256) void k_softmax(
    const float* __restrict__ s1, const float* __restrict__ s2,
    const float* __restrict__ L, _Float16* __restrict__ alpha) {
  __shared__ float ebuf[NNODES];
  __shared__ float red[256];
  const int row = blockIdx.x;
  const int t = threadIdx.x;
  const float s1v = s1[row];
  const float* Lr = L + (size_t)row * NNODES;

  float lmax = -3.4e38f;
  for (int j = t; j < NNODES; j += 256) {
    const float lv = Lr[j];
    float e;
    if (lv != 0.0f) {
      const float v = s1v + s2[j];
      e = (v >= 0.0f) ? v : SLOPE * v;
    } else {
      e = NEGV;
    }
    ebuf[j] = e;
    lmax = fmaxf(lmax, e);
  }
  red[t] = lmax;
  __syncthreads();
  for (int s = 128; s > 0; s >>= 1) {
    if (t < s) red[t] = fmaxf(red[t], red[t + s]);
    __syncthreads();
  }
  const float m = red[0];
  __syncthreads();

  float lsum = 0.f;
  for (int j = t; j < NNODES; j += 256) {
    const float ex = expf(ebuf[j] - m);
    ebuf[j] = ex;
    lsum += ex;
  }
  red[t] = lsum;
  __syncthreads();
  for (int s = 128; s > 0; s >>= 1) {
    if (t < s) red[t] += red[t + s];
    __syncthreads();
  }
  const float inv = 1.0f / red[0];
  __syncthreads();

  _Float16* ar = alpha + (size_t)row * NNODES;
  for (int j = t; j < NNODES; j += 256) ar[j] = (_Float16)(ebuf[j] * inv);
}

// ---------------------------------------------------------------------------
// k_gemm: C[M x NDIM] (+)= A[M x KDIM] @ B[KDIM x NDIM], row-major, f16 WMMA.
//   A: f16, or f32 converted while staging (A_F32).
//   B: always f16 with row stride NDIM.
//   C: f16 store (OUT_HALF) or f32 store / f32 accumulate (ACCUM); row stride LDC.
// Block tile 128x128, BK=32, 256 threads = 8 wave32 waves; per wave a 64x32
// sub-tile = 4x2 v_wmma_f32_16x16x32_f16 accumulators. Tile fills are 128-bit,
// software-pipelined through registers (fetch k+32 while computing k).
// ---------------------------------------------------------------------------
template <bool A_F32, bool OUT_HALF, bool ACCUM, int NDIM, int KDIM, int LDC>
__global__ __launch_bounds__(256) void k_gemm(
    const void* __restrict__ Ap, const _Float16* __restrict__ B,
    void* __restrict__ Cp) {
  __shared__ _Float16 sA[128][40];  // [m][k], 80B row -> 16B-aligned b128 stores
  __shared__ _Float16 sB[128][40];  // [n][k] (B tile transposed)

  const int t = threadIdx.x;
  const int lane = t & 31;
  const int wave = t >> 5;
  const int wm = wave >> 2;   // 0..1 -> 64 rows
  const int wn = wave & 3;    // 0..3 -> 32 cols
  const int bm = blockIdx.y * 128;
  const int bn = blockIdx.x * 128;

  // A staging: thread owns rows r0 and r0+64, 8 halves at col c0
  const int r0 = t >> 2;
  const int c0 = (t & 3) << 3;
  // B staging: thread owns column n, 16 k-values starting at kb
  const int nB = t & 127;
  const int kb = (t >> 7) << 4;

  v8h stA[2], stB[2];

  auto fetch = [&](int kt) {
    if constexpr (A_F32) {
      const float* A = (const float*)Ap;
#pragma unroll
      for (int h = 0; h < 2; ++h) {
        const float* p = A + (size_t)(bm + r0 + 64 * h) * KDIM + kt + c0;
        const f4 a0 = ((const f4*)p)[0];
        const f4 a1 = ((const f4*)p)[1];
        v8h hv;
        hv[0] = (_Float16)a0.x; hv[1] = (_Float16)a0.y;
        hv[2] = (_Float16)a0.z; hv[3] = (_Float16)a0.w;
        hv[4] = (_Float16)a1.x; hv[5] = (_Float16)a1.y;
        hv[6] = (_Float16)a1.z; hv[7] = (_Float16)a1.w;
        stA[h] = hv;
      }
    } else {
      const _Float16* A = (const _Float16*)Ap;
#pragma unroll
      for (int h = 0; h < 2; ++h)
        stA[h] = *(const v8h*)(A + (size_t)(bm + r0 + 64 * h) * KDIM + kt + c0);
    }
    // B: coalesced across lanes (consecutive n), 16 strided rows per thread
#pragma unroll
    for (int i = 0; i < 16; ++i)
      stB[i >> 3][i & 7] = B[(size_t)(kt + kb + i) * NDIM + bn + nB];
    // prefetch 2 tiles ahead into L2 (global_prefetch_b8)
    if (kt + 64 < KDIM) {
      if constexpr (A_F32)
        __builtin_prefetch((const float*)Ap + (size_t)(bm + r0) * KDIM + kt + 64 + c0, 0, 1);
      else
        __builtin_prefetch((const _Float16*)Ap + (size_t)(bm + r0) * KDIM + kt + 64 + c0, 0, 1);
      __builtin_prefetch(B + (size_t)(kt + 64 + kb) * NDIM + bn + nB, 0, 1);
    }
  };

  v8f acc[4][2] = {};
  const int rrow = lane & 15;
  const int kh   = lane >> 4;

  fetch(0);

  for (int kt = 0; kt < KDIM; kt += 32) {
    // commit staged tile to LDS
    *(v8h*)&sA[r0][c0]      = stA[0];
    *(v8h*)&sA[r0 + 64][c0] = stA[1];
    *(v8h*)&sB[nB][kb]      = stB[0];
    *(v8h*)&sB[nB][kb + 8]  = stB[1];
    __syncthreads();

    if (kt + 32 < KDIM) fetch(kt + 32);  // overlaps with fragment loads + WMMA

    v16h af[4], bf[2];
#pragma unroll
    for (int mt = 0; mt < 4; ++mt) {
      const int r = wm * 64 + mt * 16 + rrow;
#pragma unroll
      for (int v = 0; v < 8; ++v) {
        // A 16x32 f16 layout: lane group 0 -> K {0-7,16-23}; group 1 -> {8-15,24-31}
        const int kk = ((v >> 2) << 4) + (kh << 3) + ((v & 3) << 1);
        af[mt][2 * v]     = sA[r][kk];
        af[mt][2 * v + 1] = sA[r][kk + 1];
      }
    }
#pragma unroll
    for (int nt = 0; nt < 2; ++nt) {
      const int n = wn * 32 + nt * 16 + rrow;
#pragma unroll
      for (int v = 0; v < 8; ++v) {
        // B 32x16 f16 layout: lane group 0 -> K {0-7,16-23}; group 1 -> {8-15,24-31}
        const int kk = ((v >> 2) << 4) + (kh << 3) + ((v & 3) << 1);
        bf[nt][2 * v]     = sB[n][kk];
        bf[nt][2 * v + 1] = sB[n][kk + 1];
      }
    }

#pragma unroll
    for (int mt = 0; mt < 4; ++mt)
#pragma unroll
      for (int nt = 0; nt < 2; ++nt)
        acc[mt][nt] = __builtin_amdgcn_wmma_f32_16x16x32_f16(
            false, af[mt], false, bf[nt], (short)0, acc[mt][nt], false, false);

    __syncthreads();
  }

  // epilogue: D layout: VGPR r -> M = r + 8*(lane>=16), N = lane&15
  const int nloc = lane & 15;
  const int mhi  = (lane >> 4) << 3;
#pragma unroll
  for (int mt = 0; mt < 4; ++mt) {
#pragma unroll
    for (int nt = 0; nt < 2; ++nt) {
#pragma unroll
      for (int r = 0; r < 8; ++r) {
        const int m = bm + wm * 64 + mt * 16 + mhi + r;
        const int n = bn + wn * 32 + nt * 16 + nloc;
        const float v = acc[mt][nt][r];
        if constexpr (OUT_HALF) {
          ((_Float16*)Cp)[(size_t)m * LDC + n] = (_Float16)v;
        } else if constexpr (ACCUM) {
          ((float*)Cp)[(size_t)m * LDC + n] += v;
        } else {
          ((float*)Cp)[(size_t)m * LDC + n] = v;
        }
      }
    }
  }
}

// ---------------------------------------------------------------------------
// k_pack_w: W_all (640x128, f16) = [Wi_w[0]; Wi_w[1]; Ws_w[0]; Ws_w[1]; Wh_w]
// ---------------------------------------------------------------------------
__global__ __launch_bounds__(256) void k_pack_w(
    const float* __restrict__ Wi_w, const float* __restrict__ Ws_w,
    const float* __restrict__ Wh_w, _Float16* __restrict__ W_all) {
  const int idx = blockIdx.x * 256 + threadIdx.x;  // 640*128
  const int k = idx >> 7, o = idx & 127;
  float v;
  if (k < 256)      v = Wi_w[idx];
  else if (k < 512) v = Ws_w[(size_t)(k - 256) * 128 + o];
  else              v = Wh_w[(size_t)(k - 512) * 128 + o];
  W_all[idx] = (_Float16)v;
}

// ---------------------------------------------------------------------------
// k_bias: z[i,o] = Wi_b[0,o]+Wi_b[1,o]+Ws_b[0,o]+Ws_b[1,o]+Wh_b[o]
// ---------------------------------------------------------------------------
__global__ __launch_bounds__(256) void k_bias(
    const float* __restrict__ Wi_b, const float* __restrict__ Ws_b,
    const float* __restrict__ Wh_b, float* __restrict__ z) {
  const int idx = blockIdx.x * 256 + threadIdx.x;
  const int o = idx & (COUTC - 1);
  z[idx] = Wi_b[o] + Wi_b[COUTC + o] + Ws_b[o] + Ws_b[COUTC + o] + Wh_b[o];
}

// ---------------------------------------------------------------------------
// Host launch
// ---------------------------------------------------------------------------
extern "C" void kernel_launch(void* const* d_in, const int* in_sizes, int n_in,
                              void* d_out, int out_size, void* d_ws, size_t ws_size,
                              hipStream_t stream) {
  const float* x       = (const float*)d_in[0];
  const float* Lu      = (const float*)d_in[1];
  const float* Ld      = (const float*)d_in[2];
  const float* P       = (const float*)d_in[3];
  const float* Wi_w    = (const float*)d_in[4];
  const float* Wi_b    = (const float*)d_in[5];
  const float* Ws_w    = (const float*)d_in[6];
  const float* Ws_b    = (const float*)d_in[7];
  const float* Wh_w    = (const float*)d_in[8];
  const float* Wh_b    = (const float*)d_in[9];
  const float* att_irr = (const float*)d_in[10];
  const float* att_sol = (const float*)d_in[11];
  float* z = (float*)d_out;

  // workspace carve-up (~102.4 MiB)
  uint8_t* w = (uint8_t*)d_ws;
  _Float16* xh   = (_Float16*)w; w += (size_t)NNODES * CINC * 2;       // 1 MiB
  float*    vbuf = (float*)w;    w += 1024 * 4;
  float*    sbuf = (float*)w;    w += (size_t)4 * NNODES * 4;
  _Float16* Ah   = (_Float16*)w; w += (size_t)NNODES * NNODES * 2;     // 32 MiB
  _Float16* A1h  = (_Float16*)w; w += (size_t)NNODES * NNODES * 2;     // 32 MiB
  _Float16* Lh   = (_Float16*)w; w += (size_t)NNODES * NNODES * 2;     // 32 MiB
  _Float16* Tall = (_Float16*)w; w += (size_t)NNODES * 640 * 2;        // 5 MiB
  _Float16* Wall = (_Float16*)w; w += (size_t)640 * 128 * 2;

  const float* s_irr1 = sbuf;
  const float* s_irr2 = sbuf + NNODES;
  const float* s_sol1 = sbuf + 2 * NNODES;
  const float* s_sol2 = sbuf + 3 * NNODES;

  // ---- prep ----
  k_prep<<<1, 256, 0, stream>>>(Wi_w, Wi_b, Ws_w, Ws_b, att_irr, att_sol, vbuf);
  k_scores<<<NNODES / 256, 256, 0, stream>>>(x, vbuf, sbuf);
  k_cvt<<<(NNODES * CINC / 4) / 256, 256, 0, stream>>>(x, xh, NNODES * CINC / 4);
  k_pack_w<<<(640 * 128) / 256, 256, 0, stream>>>(Wi_w, Ws_w, Wh_w, Wall);

  const dim3 g_nx(1, NNODES / 128);             // alpha @ x  (N=128)
  const dim3 g_nn(NNODES / 128, NNODES / 128);  // alpha @ L  (N=4096)
  constexpr int NSQ = NNODES * NNODES / 4;

  // ---- irrotational branch ----
  k_cvt<<<NSQ / 256, 256, 0, stream>>>(Ld, Lh, NSQ);
  k_softmax<<<NNODES, 256, 0, stream>>>(s_irr1, s_irr2, Ld, Ah);
  k_gemm<false, true, false, 128, NNODES, 640>
      <<<g_nx, 256, 0, stream>>>(Ah, xh, Tall + 0);
  k_gemm<false, true, false, NNODES, NNODES, NNODES>
      <<<g_nn, 256, 0, stream>>>(Ah, Lh, A1h);
  k_gemm<false, true, false, 128, NNODES, 640>
      <<<g_nx, 256, 0, stream>>>(A1h, xh, Tall + 128);

  // ---- solenoidal branch (reuses Ah / A1h / Lh) ----
  k_cvt<<<NSQ / 256, 256, 0, stream>>>(Lu, Lh, NSQ);
  k_softmax<<<NNODES, 256, 0, stream>>>(s_sol1, s_sol2, Lu, Ah);
  k_gemm<false, true, false, 128, NNODES, 640>
      <<<g_nx, 256, 0, stream>>>(Ah, xh, Tall + 256);
  k_gemm<false, true, false, NNODES, NNODES, NNODES>
      <<<g_nn, 256, 0, stream>>>(Ah, Lh, A1h);
  k_gemm<false, true, false, 128, NNODES, 640>
      <<<g_nx, 256, 0, stream>>>(A1h, xh, Tall + 384);

  // ---- harmonic projector: T4 = (f16)(P @ x) ----
  k_gemm<true, true, false, 128, NNODES, 640>
      <<<g_nx, 256, 0, stream>>>(P, xh, Tall + 512);

  // ---- epilogue: z = biases + T_all(4096x640) @ W_all(640x128) ----
  k_bias<<<(NNODES * COUTC) / 256, 256, 0, stream>>>(Wi_b, Ws_b, Wh_b, z);
  k_gemm<false, false, true, 128, 640, 128>
      <<<g_nx, 256, 0, stream>>>(Tall, Wall, z);
}
